// WindowAttention3D_1529008358018
// MI455X (gfx1250) — compile-verified, matrix operand order checked by reading, not run on hardware
//
#include <hip/hip_runtime.h>

// ---------------- static config (matches reference) ----------------
#define DIM_   192
#define HEADS_ 6
#define HD     32
#define NWT    2
#define NWX    12
#define NWY    12
#define NWIN   288          // 2*12*12
#define NTOK   256          // 4*8*8 tokens per window
#define SCALE_ 0.17677669529663687f   // 32^-0.5

typedef __attribute__((ext_vector_type(16))) _Float16 v16h;
typedef __attribute__((ext_vector_type(8)))  _Float16 v8h;
typedef __attribute__((ext_vector_type(4)))  _Float16 v4h;
typedef __attribute__((ext_vector_type(8)))  float    v8f;
typedef __attribute__((ext_vector_type(4)))  float    v4f;

// Map (window, token-in-window) -> original (pre-roll) flat token index.
// roll(x,-s) then windowize == gather at (coord + s) mod N; the final
// roll(out,+s) scatters back to exactly the same indices.
__device__ __forceinline__ int tok_orig_index(int w, int i) {
  int wti = w / (NWX * NWY);
  int rem = w - wti * (NWX * NWY);
  int wxi = rem / NWY;
  int wyi = rem - wxi * NWY;
  int tt = i >> 6, tx = (i >> 3) & 7, ty = i & 7;
  int t  = wti * 4 + tt;
  int gx = wxi * 8 + tx;
  int gy = wyi * 8 + ty;
  int ot = (t + 2) & 7;
  int ox = gx + 4; if (ox >= 96) ox -= 96;
  int oy = gy + 4; if (oy >= 96) oy -= 96;
  return (ot * 96 + ox) * 96 + oy;
}

// Load a 16x32 f16 A-fragment (also used for transposed-layout B operands).
// Per ISA 7.12.2: lanes 0-15 hold k=[0..7],[16..23], lanes 16-31 k=[8..15],[24..31],
// row m = lane%16. Two aligned 16-byte chunks per lane.
__device__ __forceinline__ v16h load_frag16(const _Float16* base, int stride,
                                            int k0, int lane) {
  int m  = lane & 15;
  int kb = (lane < 16) ? 0 : 8;
  const _Float16* p = base + m * stride + k0 + kb;
  union { v16h v; v8h h[2]; } u;
  u.h[0] = *(const v8h*)p;
  u.h[1] = *(const v8h*)(p + 16);
  return u.v;
}

__device__ __forceinline__ v8f wmma_f16(v16h a, v16h b, v8f c) {
  return __builtin_amdgcn_wmma_f32_16x16x32_f16(false, a, false, b,
                                                (short)0, c, false, false);
}

// =================== kernel 1: fused roll+partition+QKV GEMM ===================
// grid: 576 (one WG per half-window of 128 tokens), block: 256 (8 waves)
__global__ __launch_bounds__(256) void k_qkv(const float* __restrict__ x,
                                             const float* __restrict__ wqkv,
                                             _Float16* __restrict__ qkv) {
  __shared__ _Float16 Xl[128 * 200];   // 128 token rows, stride 200 (pad, 16B aligned)
  __shared__ _Float16 Wl[32 * 200];    // W^T chunk: [ncol][k]

  const int tid  = threadIdx.x;
  const int lane = tid & 31;
  const int wv   = tid >> 5;
  const int wq   = blockIdx.x >> 1;
  const int half = blockIdx.x & 1;

  // gather 128 rolled token rows, fp32 -> f16
  {
    int row = tid >> 1;
    int c0  = (tid & 1) * 96;
    int i   = half * 128 + row;
    int tok = tok_orig_index(wq, i);
    const float* src = x + (size_t)tok * DIM_ + c0;
    _Float16* dst = Xl + row * 200 + c0;
#pragma unroll
    for (int j = 0; j < 96; j += 4) {
      v4f f = *(const v4f*)(src + j);
      *(v4h*)(dst + j) = __builtin_convertvector(f, v4h);
    }
  }
  __syncthreads();

  // A fragments are chunk-invariant: preload 6 k-steps of this wave's 16 rows
  const int mt = wv;
  v16h a[6];
#pragma unroll
  for (int kk = 0; kk < 6; ++kk)
    a[kk] = load_frag16(Xl + mt * 16 * 200, 200, kk * 32, lane);

  const int hi8 = (lane < 16) ? 0 : 8;
  const size_t plane = (size_t)HEADS_ * NWIN * NTOK * HD;  // Q/K/V plane size

  for (int chunk = 0; chunk < 18; ++chunk) {
    const int n0 = chunk * 32;
    __syncthreads();   // previous chunk's compute done before Wl overwrite
    for (int idx = tid; idx < 32 * 192; idx += 256) {
      int nc = idx & 31, k = idx >> 5;              // consecutive nc -> coalesced
      Wl[nc * 200 + k] = (_Float16)wqkv[k * 576 + n0 + nc];
    }
    __syncthreads();

#pragma unroll
    for (int nt = 0; nt < 2; ++nt) {
      v8f acc = {};
#pragma unroll
      for (int kk = 0; kk < 6; ++kk) {
        v16h b = load_frag16(Wl + nt * 16 * 200, 200, kk * 32, lane);
        acc = wmma_f16(a[kk], b, acc);
      }
      // scatter to partitioned [which][h][w][i][d] f16 layout
      int g     = n0 + nt * 16 + (lane & 15);
      int which = g / 192;                  // 0=Q 1=K 2=V (tile never straddles)
      int gl    = g - which * 192;
      int h = gl >> 5, d = gl & 31;
      float sc = (which == 0) ? SCALE_ : 1.0f;
      _Float16* dstb = qkv + (size_t)which * plane +
                       (((size_t)h * NWIN + wq) * NTOK) * HD + d;
      int ibase = half * 128 + mt * 16 + hi8;
#pragma unroll
      for (int r = 0; r < 8; ++r)
        dstb[(size_t)(ibase + r) * HD] = (_Float16)(acc[r] * sc);
    }
  }
}

// =================== kernel 2: windowed attention (per window,head) ===================
// grid: (288, 6), block: 256 (8 waves; each wave owns 32 query rows)
__global__ __launch_bounds__(256) void k_attn(const _Float16* __restrict__ qkv,
                                              _Float16* __restrict__ og) {
  __shared__ _Float16 Kl[256 * 40];        // K as [j][d], padded
  __shared__ _Float16 Vt[32 * 264];        // V^T as [d][j], padded
  __shared__ _Float16 Pl[8 * 16 * 72];     // per-wave P staging (quarter-j)

  const int tid = threadIdx.x, lane = tid & 31, wv = tid >> 5;
  const int wq = blockIdx.x, h = blockIdx.y;
  const size_t plane = (size_t)HEADS_ * NWIN * NTOK * HD;
  const size_t base  = (((size_t)h * NWIN + wq) * NTOK) * HD;
  const _Float16* Qg = qkv + base;
  const _Float16* Kg = qkv + plane + base;
  const _Float16* Vg = qkv + 2 * plane + base;

  { // cooperative K load + V transpose
    const _Float16* ks = Kg + tid * HD;
    _Float16* kd = Kl + tid * 40;
#pragma unroll
    for (int j = 0; j < 32; j += 8) *(v8h*)(kd + j) = *(const v8h*)(ks + j);
    const _Float16* vs = Vg + tid * HD;
#pragma unroll
    for (int d = 0; d < 32; ++d) Vt[d * 264 + tid] = vs[d];
  }
  __syncthreads();

  const bool maskT = (wq >= 144);                 // wt index == 1 (last)
  const int  rem   = wq % 144;
  const bool maskX = (rem / 12) == 11;
  const bool maskY = (rem % 12) == 11;
  const bool anymask = maskT | maskX | maskY;
  const int hi8 = (lane < 16) ? 0 : 8;

  for (int mt2 = 0; mt2 < 2; ++mt2) {
    const int mbase = wv * 32 + mt2 * 16;

    // Q fragment straight from global (partitioned layout, 16B aligned)
    v16h qf;
    {
      int m  = mbase + (lane & 15);
      int kb = (lane < 16) ? 0 : 8;
      const _Float16* p = Qg + m * HD + kb;
      union { v16h v; v8h hh[2]; } u;
      u.hh[0] = *(const v8h*)p;
      u.hh[1] = *(const v8h*)(p + 16);
      qf = u.v;
    }

    // S = Q K^T : 16 tiles of 16x16 kept in registers
    v8f s[16];
#pragma unroll
    for (int jt = 0; jt < 16; ++jt) {
      v16h bf = load_frag16(Kl + jt * 16 * 40, 40, 0, lane);
      v8f z = {};
      s[jt] = wmma_f16(qf, bf, z);
    }

    if (anymask) {
#pragma unroll
      for (int jt = 0; jt < 16; ++jt) {
        int j = jt * 16 + (lane & 15);
        bool fjt = (j >> 6) >= 2;
        bool fjx = ((j >> 3) & 7) >= 4;
        bool fjy = (j & 7) >= 4;
#pragma unroll
        for (int r = 0; r < 8; ++r) {
          int i = mbase + r + hi8;
          bool fit = (i >> 6) >= 2;
          bool fix = ((i >> 3) & 7) >= 4;
          bool fiy = (i & 7) >= 4;
          bool bad = (maskT && (fit != fjt)) || (maskX && (fix != fjx)) ||
                     (maskY && (fiy != fjy));
          if (bad) s[jt][r] = -1e30f;
        }
      }
    }

    // row softmax: each row lives in one 16-lane half; xor tree reduces both halves
    float rinv[8];
#pragma unroll
    for (int r = 0; r < 8; ++r) {
      float m = -1e30f;
#pragma unroll
      for (int jt = 0; jt < 16; ++jt) m = fmaxf(m, s[jt][r]);
#pragma unroll
      for (int off = 1; off < 16; off <<= 1) m = fmaxf(m, __shfl_xor(m, off, 32));
      float sum = 0.f;
#pragma unroll
      for (int jt = 0; jt < 16; ++jt) {
        float p = __expf(s[jt][r] - m);
        s[jt][r] = p;
        sum += p;
      }
#pragma unroll
      for (int off = 1; off < 16; off <<= 1) sum += __shfl_xor(sum, off, 32);
      rinv[r] = 1.0f / sum;
    }

    // O = P V via quarter-j staging through per-wave LDS (C-layout -> A-layout)
    v8f o0 = {}, o1 = {};
    _Float16* Pw = Pl + wv * 16 * 72;
#pragma unroll
    for (int q = 0; q < 4; ++q) {
#pragma unroll
      for (int jq = 0; jq < 4; ++jq) {
        int jt = q * 4 + jq;
#pragma unroll
        for (int r = 0; r < 8; ++r)
          Pw[(r + hi8) * 72 + jq * 16 + (lane & 15)] = (_Float16)s[jt][r];
      }
      // wave-private LDS slice: DS ops from one wave stay in order
#pragma unroll
      for (int kq = 0; kq < 2; ++kq) {
        v16h pa = load_frag16(Pw, 72, kq * 32, lane);
        v16h b0 = load_frag16(Vt,             264, q * 64 + kq * 32, lane);
        v16h b1 = load_frag16(Vt + 16 * 264,  264, q * 64 + kq * 32, lane);
        o0 = wmma_f16(pa, b0, o0);
        o1 = wmma_f16(pa, b1, o1);
      }
    }

    // write merged-channel f16 O: [w*256+i][h*32+d]
    _Float16* dst = og + ((size_t)wq * NTOK) * DIM_ + h * HD;
#pragma unroll
    for (int r = 0; r < 8; ++r) {
      int i = mbase + r + hi8;
      int n = lane & 15;
      dst[(size_t)i * DIM_ + n]      = (_Float16)(o0[r] * rinv[r]);
      dst[(size_t)i * DIM_ + 16 + n] = (_Float16)(o1[r] * rinv[r]);
    }
  }
}

// =================== kernel 3: out proj + bias + inverse roll scatter ===================
// grid: 576 (half-window per WG), block: 256
__global__ __launch_bounds__(256) void k_out(const _Float16* __restrict__ og,
                                             const float* __restrict__ wout,
                                             const float* __restrict__ bout,
                                             float* __restrict__ out) {
  __shared__ _Float16 Xl[128 * 200];
  __shared__ _Float16 Wl[32 * 200];

  const int tid = threadIdx.x, lane = tid & 31, wv = tid >> 5;
  const int wq = blockIdx.x >> 1, half = blockIdx.x & 1;

  {
    int row = tid >> 1, c0 = (tid & 1) * 96;
    int i = half * 128 + row;
    const _Float16* src = og + ((size_t)wq * NTOK + i) * DIM_ + c0;
    _Float16* dst = Xl + row * 200 + c0;
#pragma unroll
    for (int j = 0; j < 96; j += 8) *(v8h*)(dst + j) = *(const v8h*)(src + j);
  }
  __syncthreads();

  const int mt = wv;
  v16h a[6];
#pragma unroll
  for (int kk = 0; kk < 6; ++kk)
    a[kk] = load_frag16(Xl + mt * 16 * 200, 200, kk * 32, lane);

  const int hi8 = (lane < 16) ? 0 : 8;

  for (int chunk = 0; chunk < 6; ++chunk) {
    const int n0 = chunk * 32;
    __syncthreads();
    for (int idx = tid; idx < 32 * 192; idx += 256) {
      int nc = idx & 31, k = idx >> 5;
      Wl[nc * 200 + k] = (_Float16)wout[k * 192 + n0 + nc];
    }
    __syncthreads();

#pragma unroll
    for (int nt = 0; nt < 2; ++nt) {
      v8f acc = {};
#pragma unroll
      for (int kk = 0; kk < 6; ++kk) {
        v16h b = load_frag16(Wl + nt * 16 * 200, 200, kk * 32, lane);
        acc = wmma_f16(a[kk], b, acc);
      }
      int g = n0 + nt * 16 + (lane & 15);
      float bias = bout[g];
#pragma unroll
      for (int r = 0; r < 8; ++r) {
        int i = half * 128 + mt * 16 + r + hi8;
        int tok = tok_orig_index(wq, i);
        out[(size_t)tok * DIM_ + g] = acc[r] + bias;
      }
    }
  }
}

// ======================= launcher =======================
extern "C" void kernel_launch(void* const* d_in, const int* in_sizes, int n_in,
                              void* d_out, int out_size, void* d_ws, size_t ws_size,
                              hipStream_t stream) {
  (void)in_sizes; (void)n_in; (void)out_size; (void)ws_size;
  const float* x     = (const float*)d_in[0];
  const float* wqkv  = (const float*)d_in[1];
  const float* wout  = (const float*)d_in[2];
  const float* bout  = (const float*)d_in[3];
  float* out = (float*)d_out;

  // workspace: QKV planes (f16) then merged O (f16)
  const size_t qkv_elems = 3 * (size_t)HEADS_ * NWIN * NTOK * HD; // 42,467,328
  _Float16* qkvbuf = (_Float16*)d_ws;
  _Float16* ogbuf  = qkvbuf + qkv_elems;

  k_qkv <<<dim3(2 * NWIN),     256, 0, stream>>>(x, wqkv, qkvbuf);
  k_attn<<<dim3(NWIN, HEADS_), 256, 0, stream>>>(qkvbuf, ogbuf);
  k_out <<<dim3(2 * NWIN),     256, 0, stream>>>(ogbuf, wout, bout, out);
}
